// PatchedInternLMAttention_59897613910435
// MI455X (gfx1250) — matrix operand order repaired and checked by reading, not compiled, not run on hardware
//
#include <hip/hip_runtime.h>
#include <hip/hip_bf16.h>

#define HEADS    32
#define HEAD_DIM 128
#define HID      4096
#define TTOK     3072
#define NB       4
#define MAXB     24
#define MAXKV    1536
#define KVBLK    64

typedef __attribute__((ext_vector_type(16))) __bf16 v16bf;
typedef __attribute__((ext_vector_type(8)))  float  v8f;
typedef __attribute__((ext_vector_type(4)))  unsigned int v4u;

union FragBF {
    v16bf v;
    v4u   u[2];
    unsigned short s[16];
};

__device__ __forceinline__ unsigned short f2bf(float f) {
    unsigned u = __float_as_uint(f);
    u += 0x7FFFu + ((u >> 16) & 1u);        // round-to-nearest-even
    return (unsigned short)(u >> 16);
}

__device__ __forceinline__ void cvt8(const float* __restrict__ p, unsigned short* dst) {
    float4 a = *(const float4*)p;
    float4 b = *(const float4*)(p + 4);
    dst[0] = f2bf(a.x); dst[1] = f2bf(a.y); dst[2] = f2bf(a.z); dst[3] = f2bf(a.w);
    dst[4] = f2bf(b.x); dst[5] = f2bf(b.y); dst[6] = f2bf(b.z); dst[7] = f2bf(b.w);
}

// ---------------------------------------------------------------- conversions
__global__ void cvt_f32_to_bf16(const float* __restrict__ src,
                                unsigned short* __restrict__ dst, long n) {
    long i = (long)blockIdx.x * blockDim.x + threadIdx.x;
    long stride = (long)gridDim.x * blockDim.x;
    for (; i < n; i += stride) dst[i] = f2bf(src[i]);
}

// ---------------------------------------------------------------- RoPE (q,k in place)
__global__ void rope_qk(float* __restrict__ q, float* __restrict__ k,
                        const int* __restrict__ pos, int n) {
    int i = blockIdx.x * blockDim.x + threadIdx.x;
    if (i >= n) return;                     // n = TTOK*HEADS*64
    int j = i & 63;
    int h = (i >> 6) & (HEADS - 1);
    int t = i >> 11;
    float inv = __powf(10000.f, -(float)(2 * j) * (1.0f / 128.f));
    float f = (float)pos[t] * inv;
    float s, c;
    __sincosf(f, &s, &c);
    size_t base = (size_t)t * HID + h * HEAD_DIM + j;
    float x1 = q[base], x2 = q[base + 64];
    q[base]      = x1 * c - x2 * s;
    q[base + 64] = x2 * c + x1 * s;
    x1 = k[base]; x2 = k[base + 64];
    k[base]      = x1 * c - x2 * s;
    k[base + 64] = x2 * c + x1 * s;
}

// ---------------------------------------------------------------- gather paged cache + fresh -> padded bf16
__global__ void kv_pad(const float* __restrict__ cache, const float* __restrict__ fresh,
                       const int* __restrict__ blkoff, const int* __restrict__ qstart,
                       const int* __restrict__ qlens,  const int* __restrict__ kvlens,
                       unsigned short* __restrict__ pad, long n) {
    long i = (long)blockIdx.x * blockDim.x + threadIdx.x;
    long stride = (long)gridDim.x * blockDim.x;
    for (; i < n; i += stride) {
        int d = (int)(i & (HEAD_DIM - 1));
        int h = (int)((i >> 7) & (HEADS - 1));
        long r = i >> 12;
        int j = (int)(r % MAXKV);
        int b = (int)(r / MAXKV);
        int kvl = kvlens[b], ql = qlens[b];
        int hist = kvl - ql;
        float val = 0.f;
        if (j < hist) {
            int blk = blkoff[b * MAXB + (j >> 6)];
            val = cache[(((size_t)blk * KVBLK + (j & 63)) * HEADS + h) * HEAD_DIM + d];
        } else if (j < kvl) {
            val = fresh[(size_t)(qstart[b] + j - hist) * HID + h * HEAD_DIM + d];
        }
        pad[i] = f2bf(val);
    }
}

// ---------------------------------------------------------------- bf16 WMMA GEMM:  C[M,N] = A[M,K] * W[N,K]^T + bias
// block = 128 threads (4 waves); wave w computes rows m0+16w, 64 columns
__global__ __launch_bounds__(128) void gemm_bf16(const unsigned short* __restrict__ A,
                                                 const unsigned short* __restrict__ W,
                                                 const float* __restrict__ bias,
                                                 float* __restrict__ C,
                                                 int M, int N, int K) {
    int lane = threadIdx.x & 31;
    int wave = threadIdx.x >> 5;
    int l    = lane & 15;
    int halfw = lane >> 4;
    int m0 = blockIdx.y * 64 + wave * 16;
    int n0 = blockIdx.x * 64;

    const unsigned short* arow = A + (size_t)(m0 + l) * K;
    int aoff0 = halfw * 8;          // A layout: half0 K{0..7,16..23}, half1 K{8..15,24..31}
    int aoff1 = 16 + halfw * 8;

    v8f acc[4] = {};
    for (int k0 = 0; k0 < K; k0 += 32) {
        __builtin_prefetch((const void*)(arow + k0 + 256), 0, 1);
        FragBF a;
        a.u[0] = *(const v4u*)(arow + k0 + aoff0);
        a.u[1] = *(const v4u*)(arow + k0 + aoff1);
#pragma unroll
        for (int t = 0; t < 4; ++t) {
            // B layout: lane l holds column n0+t*16+l, half0 K{0..7,16..23}, half1 K{8..15,24..31}
            const unsigned short* wrow = W + (size_t)(n0 + t * 16 + l) * K + k0 + halfw * 8;
            FragBF bfr;
            bfr.u[0] = *(const v4u*)(wrow);
            bfr.u[1] = *(const v4u*)(wrow + 16);
            acc[t] = __builtin_amdgcn_wmma_f32_16x16x32_bf16(
                false, a.v, false, bfr.v, (short)0, acc[t], false, false);
        }
    }
#pragma unroll
    for (int t = 0; t < 4; ++t) {
        int n = n0 + t * 16 + l;
        float bb = bias[n];
#pragma unroll
        for (int v = 0; v < 8; ++v) {
            int m = m0 + halfw * 8 + v;      // D layout: lanes0-15 M=v, lanes16-31 M=8+v
            C[(size_t)m * N + n] = acc[t][v] + bb;
        }
    }
}

// ---------------------------------------------------------------- flash attention, one wave per (q-tile16, head)
__global__ __launch_bounds__(32) void flash_attn(const float* __restrict__ q,
                                                 const unsigned short* __restrict__ kpad,
                                                 const unsigned short* __restrict__ vpad,
                                                 const int* __restrict__ qstart,
                                                 const int* __restrict__ qlens,
                                                 const int* __restrict__ kvlens,
                                                 float* __restrict__ out) {
    __shared__ unsigned short ptile[16 * 32];    // P staging for D->A transpose
    __shared__ unsigned short vtile[32 * 128];   // async-staged V tile (8KB)

    int lane  = threadIdx.x;
    int l     = lane & 15;
    int halfw = lane >> 4;
    int tok0  = blockIdx.x * 16;
    int h     = blockIdx.y;

    int b = 0;
#pragma unroll
    for (int i = 1; i < NB; ++i) if (tok0 >= qstart[i]) b = i;
    int ql = qlens[b], kvl = kvlens[b];
    int hist = kvl - ql;
    int i0 = tok0 - qstart[b];
    const float scale = 0.08838834764831845f;   // 1/sqrt(128)

    // per-lane LDS byte address of this lane's V-tile row (low 32 bits of flat addr == LDS offset)
    unsigned vrow_lds = (unsigned)(uintptr_t)(&vtile[lane * HEAD_DIM]);

    // Q fragments: 4 chunks of K=32 over d, A-matrix layout
    const float* qrow = q + (size_t)(tok0 + l) * HID + h * HEAD_DIM;
    FragBF qa[4];
#pragma unroll
    for (int c = 0; c < 4; ++c) {
        cvt8(qrow + c * 32 + halfw * 8,      &qa[c].s[0]);
        cvt8(qrow + c * 32 + 16 + halfw * 8, &qa[c].s[8]);
    }

    v8f o[8] = {};
    float mrow[8], lsum[8];
#pragma unroll
    for (int v = 0; v < 8; ++v) { mrow[v] = -3.0e38f; lsum[v] = 0.f; }

    int jmax = hist + i0 + 15;
    for (int kv0 = 0; kv0 <= jmax; kv0 += 32) {
        // ---- kick off async DMA of the 32x128 bf16 V tile into LDS (overlaps S + softmax below)
        {
            const unsigned short* vsrc =
                vpad + ((size_t)(b * MAXKV + kv0 + lane) * HEADS + h) * HEAD_DIM;
#pragma unroll
            for (int i = 0; i < 16; ++i) {
                asm volatile("global_load_async_to_lds_b128 %0, %1, off"
                             :: "v"(vrow_lds + i * 16), "v"(vsrc + i * 8)
                             : "memory");
            }
        }
        // ---- S = Q K^T  (two 16x16 tiles over kv)
        v8f s[2] = {};
#pragma unroll
        for (int t = 0; t < 2; ++t) {
            const unsigned short* krow =
                kpad + ((size_t)(b * MAXKV + kv0 + t * 16 + l) * HEADS + h) * HEAD_DIM + halfw * 8;
#pragma unroll
            for (int c = 0; c < 4; ++c) {
                FragBF kfr;
                kfr.u[0] = *(const v4u*)(krow + c * 32);
                kfr.u[1] = *(const v4u*)(krow + c * 32 + 16);
                s[t] = __builtin_amdgcn_wmma_f32_16x16x32_bf16(
                    false, qa[c].v, false, kfr.v, (short)0, s[t], false, false);
            }
        }
        // ---- mask + online softmax; write P (bf16) to LDS in D layout
#pragma unroll
        for (int v = 0; v < 8; ++v) {
            int row_i = i0 + halfw * 8 + v;
            int allow = hist + row_i;                       // key j allowed iff j <= allow
            float s0 = (kv0 + l      <= allow) ? s[0][v] * scale : -3.0e38f;
            float s1 = (kv0 + 16 + l <= allow) ? s[1][v] * scale : -3.0e38f;
            float mx = fmaxf(s0, s1);
#pragma unroll
            for (int d = 1; d < 16; d <<= 1) mx = fmaxf(mx, __shfl_xor(mx, d, 32));
            float mnew  = fmaxf(mrow[v], mx);
            float alpha = __expf(mrow[v] - mnew);
            mrow[v] = mnew;
            float p0 = __expf(s0 - mnew);
            float p1 = __expf(s1 - mnew);
            float rs = p0 + p1;
#pragma unroll
            for (int d = 1; d < 16; d <<= 1) rs += __shfl_xor(rs, d, 32);
            lsum[v] = lsum[v] * alpha + rs;
#pragma unroll
            for (int c = 0; c < 8; ++c) o[c][v] = o[c][v] * alpha;
            int prow = halfw * 8 + v;
            ptile[prow * 32 + l]      = f2bf(p0);
            ptile[prow * 32 + 16 + l] = f2bf(p1);
        }
        asm volatile("s_wait_dscnt 0" ::: "memory");
        // ---- P fragment (A-matrix layout) from LDS
        FragBF pf;
        pf.u[0] = *(const v4u*)(&ptile[l * 32 + halfw * 16]);
        pf.u[1] = *(const v4u*)(&ptile[l * 32 + halfw * 16 + 8]);
        // ---- wait for async V tile, then O += P * V  (8 d-tiles of 16)
        asm volatile("s_wait_asynccnt 0" ::: "memory");
#pragma unroll
        for (int dc = 0; dc < 8; ++dc) {
            FragBF vf;
#pragma unroll
            for (int kk = 0; kk < 16; ++kk) {
                vf.s[kk] = vtile[(halfw * 16 + kk) * HEAD_DIM + dc * 16 + l];
            }
            o[dc] = __builtin_amdgcn_wmma_f32_16x16x32_bf16(
                false, pf.v, false, vf.v, (short)0, o[dc], false, false);
        }
    }
    // ---- normalize + store
#pragma unroll
    for (int v = 0; v < 8; ++v) {
        float invl = 1.f / lsum[v];
        int tokr = tok0 + halfw * 8 + v;
#pragma unroll
        for (int dc = 0; dc < 8; ++dc) {
            out[(size_t)tokr * HID + h * HEAD_DIM + dc * 16 + l] = o[dc][v] * invl;
        }
    }
}

// ---------------------------------------------------------------- host launcher
extern "C" void kernel_launch(void* const* d_in, const int* in_sizes, int n_in,
                              void* d_out, int out_size, void* d_ws, size_t ws_size,
                              hipStream_t stream) {
    const float* hs     = (const float*)d_in[0];
    const int*   pos    = (const int*)d_in[1];
    const int*   qstart = (const int*)d_in[2];
    const int*   qlens  = (const int*)d_in[3];
    const int*   kvlens = (const int*)d_in[4];
    const int*   blkoff = (const int*)d_in[5];
    const float* pastk  = (const float*)d_in[6];
    const float* pastv  = (const float*)d_in[7];
    const float* Wq = (const float*)d_in[8];  const float* bq = (const float*)d_in[9];
    const float* Wk = (const float*)d_in[10]; const float* bk = (const float*)d_in[11];
    const float* Wv = (const float*)d_in[12]; const float* bv = (const float*)d_in[13];
    const float* Wo = (const float*)d_in[14]; const float* bo = (const float*)d_in[15];
    float* out = (float*)d_out;

    char* wp = (char*)d_ws;
    auto take = [&](size_t bytes) -> char* {
        char* p = wp;
        wp += (bytes + 255) & ~(size_t)255;
        return p;
    };
    unsigned short* hsb   = (unsigned short*)take((size_t)TTOK * HID * 2);
    unsigned short* Wqb   = (unsigned short*)take((size_t)HID * HID * 2);
    unsigned short* Wkb   = (unsigned short*)take((size_t)HID * HID * 2);
    unsigned short* Wvb   = (unsigned short*)take((size_t)HID * HID * 2);
    unsigned short* Wob   = (unsigned short*)take((size_t)HID * HID * 2);
    float*          dq    = (float*)take((size_t)TTOK * HID * 4);
    float*          dk    = (float*)take((size_t)TTOK * HID * 4);
    float*          dv    = (float*)take((size_t)TTOK * HID * 4);
    float*          attn  = (float*)take((size_t)TTOK * HID * 4);
    unsigned short* attnb = (unsigned short*)take((size_t)TTOK * HID * 2);
    unsigned short* kpadb = (unsigned short*)take((size_t)NB * MAXKV * HEADS * HEAD_DIM * 2);
    unsigned short* vpadb = (unsigned short*)take((size_t)NB * MAXKV * HEADS * HEAD_DIM * 2);

    long nhs = (long)TTOK * HID;
    long nw  = (long)HID * HID;
    cvt_f32_to_bf16<<<2048, 256, 0, stream>>>(hs, hsb, nhs);
    cvt_f32_to_bf16<<<2048, 256, 0, stream>>>(Wq, Wqb, nw);
    cvt_f32_to_bf16<<<2048, 256, 0, stream>>>(Wk, Wkb, nw);
    cvt_f32_to_bf16<<<2048, 256, 0, stream>>>(Wv, Wvb, nw);
    cvt_f32_to_bf16<<<2048, 256, 0, stream>>>(Wo, Wob, nw);

    dim3 gg(HID / 64, TTOK / 64);
    gemm_bf16<<<gg, 128, 0, stream>>>(hsb, Wqb, bq, dq, TTOK, HID, HID);
    gemm_bf16<<<gg, 128, 0, stream>>>(hsb, Wkb, bk, dk, TTOK, HID, HID);
    gemm_bf16<<<gg, 128, 0, stream>>>(hsb, Wvb, bv, dv, TTOK, HID, HID);

    int nro = TTOK * HEADS * 64;
    rope_qk<<<(nro + 255) / 256, 256, 0, stream>>>(dq, dk, pos, nro);

    long npad = (long)NB * MAXKV * HEADS * HEAD_DIM;
    kv_pad<<<2048, 256, 0, stream>>>(pastk, dk, blkoff, qstart, qlens, kvlens, kpadb, npad);
    kv_pad<<<2048, 256, 0, stream>>>(pastv, dv, blkoff, qstart, qlens, kvlens, vpadb, npad);

    dim3 ga(TTOK / 16, HEADS);
    flash_attn<<<ga, 32, 0, stream>>>(dq, kpadb, vpadb, qstart, qlens, kvlens, attn);

    cvt_f32_to_bf16<<<2048, 256, 0, stream>>>(attn, attnb, nhs);
    gemm_bf16<<<gg, 128, 0, stream>>>(attnb, Wob, bo, out, TTOK, HID, HID);
}
